// SceneQueryGSHead_19421842112786
// MI455X (gfx1250) — matrix operand order repaired and checked by reading, not compiled, not run on hardware
//
#include <hip/hip_runtime.h>
#include <math.h>

typedef __attribute__((ext_vector_type(16))) _Float16 v16h;
typedef __attribute__((ext_vector_type(8)))  float    v8f;

__device__ __forceinline__ float gelu_f(float x) {
  return 0.5f * x * (1.0f + erff(x * 0.70710678118654752f));
}

// A-matrix (16x32 f16) per-lane k index for packed pair i (0..7), half-group hs
__device__ __forceinline__ int a_kidx(int i, int hs) {
  return ((i < 4) ? (2 * i) : (16 + 2 * (i - 4))) + 8 * hs;
}

__device__ __forceinline__ v8f wmma16(v16h a, v16h b, v8f c) {
  return __builtin_amdgcn_wmma_f32_16x16x32_f16(false, a, false, b, (short)0, c,
                                                false, false);
}

// ---------------------------------------------------------------------------
// Generic fused GEMM: C[M,N] = act(A @ W + bias) (+resid). A row-major [M,K],
// W row-major [K,N]. Optional A2 broadcast-add with row divisor (proj+query).
// Block: 256 thr (8 waves), tile 64(M) x 128(N); waves 4(M) x 2(N); each wave
// computes 16x64 = 4 WMMA tiles per K-step reusing one A fragment. K staged 32
// at a time as f16 in LDS; next K tile prefetched (global_prefetch_b8).
// ---------------------------------------------------------------------------
template <int ACT>
__global__ __launch_bounds__(256) void gemm_kernel(
    const float* __restrict__ A, const float* __restrict__ A2, int a2div,
    const float* __restrict__ W, const float* __restrict__ bias,
    const float* __restrict__ resid, float* __restrict__ C,
    int M, int N, int K) {
  __shared__ _Float16 sA[64][32];
  __shared__ _Float16 sB[128][32];
  const int tid = threadIdx.x;
  const int wave = tid >> 5;
  const int lane = tid & 31;
  const int hs = lane >> 4;
  const int l16 = lane & 15;
  const int wm = wave >> 1;   // 0..3 : 16-row strip within block
  const int wn = wave & 1;    // 0..1 : 64-col strip within block
  const int row0 = blockIdx.x * 64;
  const int col0 = blockIdx.y * 128;
  v8f acc[4] = {};
  for (int k0 = 0; k0 < K; k0 += 32) {
    __syncthreads();
    // stage A tile 64x32 (f32 -> f16), fused A2 broadcast add
    for (int idx = tid; idx < 2048; idx += 256) {
      int m = idx >> 5, k = idx & 31;
      int gr = row0 + m, gk = k0 + k;
      float v = 0.0f;
      if (gr < M && gk < K) {
        v = A[(size_t)gr * K + gk];
        if (A2) v += A2[(size_t)(gr / a2div) * K + gk];
      }
      sA[m][k] = (_Float16)v;
    }
    // stage B tile 32x128 transposed so fragment k-pairs are contiguous
    for (int idx = tid; idx < 4096; idx += 256) {
      int k = idx >> 7, n = idx & 127;
      int gk = k0 + k, gn = col0 + n;
      float v = 0.0f;
      if (gk < K && gn < N) v = W[(size_t)gk * N + gn];
      sB[n][k] = (_Float16)v;
    }
    // prefetch next K tile (emits global_prefetch_b8; cacheline granularity)
    if (k0 + 32 < K) {
      int pm = row0 + (tid & 63), pka = k0 + 32 + ((tid >> 6) << 4);
      if (pm < M && pka < K) __builtin_prefetch(&A[(size_t)pm * K + pka], 0, 1);
      int pn = col0 + ((tid & 7) << 4), pkb = k0 + 32 + (tid >> 3);
      if (pkb < K && pn < N) __builtin_prefetch(&W[(size_t)pkb * N + pn], 0, 1);
    }
    __syncthreads();
    v16h af;
#pragma unroll
    for (int i = 0; i < 8; ++i) {
      int kk = a_kidx(i, hs);
      af[2 * i]     = sA[wm * 16 + l16][kk];
      af[2 * i + 1] = sA[wm * 16 + l16][kk + 1];
    }
#pragma unroll
    for (int t = 0; t < 4; ++t) {
      v16h bf;
#pragma unroll
      for (int i = 0; i < 8; ++i) {
        int bk = a_kidx(i, hs);
        bf[2 * i]     = sB[wn * 64 + t * 16 + l16][bk];
        bf[2 * i + 1] = sB[wn * 64 + t * 16 + l16][bk + 1];
      }
      acc[t] = wmma16(af, bf, acc[t]);
    }
  }
#pragma unroll
  for (int t = 0; t < 4; ++t) {
    int gn = col0 + wn * 64 + t * 16 + l16;
    if (gn < N) {
#pragma unroll
      for (int r = 0; r < 8; ++r) {
        int gm = row0 + wm * 16 + r + 8 * hs;
        if (gm < M) {
          float v = acc[t][r];
          if (bias) v += bias[gn];
          if (ACT == 1) v = gelu_f(v);
          if (resid) v += resid[(size_t)gm * N + gn];
          C[(size_t)gm * N + gn] = v;
        }
      }
    }
  }
}

// ---------------------------------------------------------------------------
// Flash attention: qkv [4096,1536] (q|k|v each 512 = 8 heads x 64).
// Grid (64, 8), block 128 (4 waves). Each wave: 16 query rows x 1 head.
// Online softmax; QK^T and PV via WMMA; P re-layouted via per-wave LDS tile.
// ---------------------------------------------------------------------------
__global__ __launch_bounds__(128) void attn_kernel(const float* __restrict__ qkv,
                                                   float* __restrict__ o_out) {
  __shared__ _Float16 sP[4][16][32];
  const int head = blockIdx.y;
  const int wave = threadIdx.x >> 5;
  const int lane = threadIdx.x & 31;
  const int hs = lane >> 4;
  const int l16 = lane & 15;
  const int r0 = blockIdx.x * 64 + wave * 16;
  const int qb = head * 64;

  v16h qa[2];
#pragma unroll
  for (int ch = 0; ch < 2; ++ch) {
#pragma unroll
    for (int i = 0; i < 8; ++i) {
      int kk = ch * 32 + a_kidx(i, hs);
      const float* qp = qkv + (size_t)(r0 + l16) * 1536 + qb + kk;
      qa[ch][2 * i]     = (_Float16)(qp[0] * 0.125f);
      qa[ch][2 * i + 1] = (_Float16)(qp[1] * 0.125f);
    }
  }
  float rmax[8], rsum[8];
  v8f oacc[4] = {};
#pragma unroll
  for (int r = 0; r < 8; ++r) { rmax[r] = -3.0e38f; rsum[r] = 0.0f; }

  for (int kb = 0; kb < 4096; kb += 32) {
    v8f s0 = {}, s1 = {};
#pragma unroll
    for (int ch = 0; ch < 2; ++ch) {
      v16h bk0, bk1;
#pragma unroll
      for (int i = 0; i < 8; ++i) {
        int kd = 512 + qb + ch * 32 + a_kidx(i, hs);
        const float* kp0 = qkv + (size_t)(kb + l16) * 1536 + kd;
        const float* kp1 = qkv + (size_t)(kb + 16 + l16) * 1536 + kd;
        bk0[2 * i] = (_Float16)kp0[0]; bk0[2 * i + 1] = (_Float16)kp0[1];
        bk1[2 * i] = (_Float16)kp1[0]; bk1[2 * i + 1] = (_Float16)kp1[1];
      }
      s0 = wmma16(qa[ch], bk0, s0);
      s1 = wmma16(qa[ch], bk1, s1);
    }
    float tmax[8], p0[8], p1[8], ps[8];
#pragma unroll
    for (int r = 0; r < 8; ++r) tmax[r] = fmaxf(s0[r], s1[r]);
#pragma unroll
    for (int off = 1; off < 16; off <<= 1)
#pragma unroll
      for (int r = 0; r < 8; ++r)
        tmax[r] = fmaxf(tmax[r], __shfl_xor(tmax[r], off, 32));
#pragma unroll
    for (int r = 0; r < 8; ++r) {
      float nm = fmaxf(rmax[r], tmax[r]);
      float f = expf(rmax[r] - nm);
      p0[r] = expf(s0[r] - nm);
      p1[r] = expf(s1[r] - nm);
      ps[r] = p0[r] + p1[r];
      rsum[r] *= f;
      rmax[r] = nm;
#pragma unroll
      for (int t = 0; t < 4; ++t) oacc[t][r] *= f;
    }
#pragma unroll
    for (int off = 1; off < 16; off <<= 1)
#pragma unroll
      for (int r = 0; r < 8; ++r) ps[r] += __shfl_xor(ps[r], off, 32);
#pragma unroll
    for (int r = 0; r < 8; ++r) rsum[r] += ps[r];

    __syncthreads();
#pragma unroll
    for (int r = 0; r < 8; ++r) {
      sP[wave][r + 8 * hs][l16]      = (_Float16)p0[r];
      sP[wave][r + 8 * hs][16 + l16] = (_Float16)p1[r];
    }
    __syncthreads();
    v16h pa;
#pragma unroll
    for (int i = 0; i < 8; ++i) {
      int kk = a_kidx(i, hs);
      pa[2 * i]     = sP[wave][l16][kk];
      pa[2 * i + 1] = sP[wave][l16][kk + 1];
    }
#pragma unroll
    for (int t = 0; t < 4; ++t) {
      v16h bv;
#pragma unroll
      for (int i = 0; i < 8; ++i) {
        int key = kb + a_kidx(i, hs);
        int col = 1024 + qb + t * 16 + l16;
        bv[2 * i]     = (_Float16)qkv[(size_t)key * 1536 + col];
        bv[2 * i + 1] = (_Float16)qkv[(size_t)(key + 1) * 1536 + col];
      }
      oacc[t] = wmma16(pa, bv, oacc[t]);
    }
  }
#pragma unroll
  for (int t = 0; t < 4; ++t)
#pragma unroll
    for (int r = 0; r < 8; ++r) {
      int row = r0 + r + 8 * hs;
      int col = qb + t * 16 + l16;
      o_out[(size_t)row * 512 + col] = oacc[t][r] / rsum[r];
    }
}

// ---------------------------------------------------------------------------
// LayerNorm (optionally followed by GELU). One block per row.
// ---------------------------------------------------------------------------
__global__ __launch_bounds__(256) void layernorm_kernel(
    const float* __restrict__ x, const float* __restrict__ w,
    const float* __restrict__ b, float* __restrict__ y, int D, int dogelu) {
  int row = blockIdx.x, tid = threadIdx.x;
  __shared__ float s1[256], s2[256];
  float a = 0.0f, a2 = 0.0f;
  for (int c = tid; c < D; c += 256) {
    float v = x[(size_t)row * D + c];
    a += v; a2 += v * v;
  }
  s1[tid] = a; s2[tid] = a2;
  __syncthreads();
  for (int st = 128; st > 0; st >>= 1) {
    if (tid < st) { s1[tid] += s1[tid + st]; s2[tid] += s2[tid + st]; }
    __syncthreads();
  }
  float mu = s1[0] / (float)D;
  float var = fmaxf(s2[0] / (float)D - mu * mu, 0.0f);
  float rstd = rsqrtf(var + 1e-5f);
  for (int c = tid; c < D; c += 256) {
    float v = (x[(size_t)row * D + c] - mu) * rstd * w[c] + b[c];
    if (dogelu) v = gelu_f(v);
    y[(size_t)row * D + c] = v;
  }
}

// ---------------------------------------------------------------------------
// Anchor statistics / encoding
// ---------------------------------------------------------------------------
__global__ __launch_bounds__(256) void center_kernel(
    const float* __restrict__ anchor, const int* __restrict__ vcp,
    float* __restrict__ stats) {
  __shared__ float sx[256], sy[256], sz[256];
  int vc = vcp[0], tid = threadIdx.x;
  float ax = 0, ay = 0, az = 0;
  for (int n = tid; n < 4096; n += 256)
    if (n < vc) { ax += anchor[n * 3]; ay += anchor[n * 3 + 1]; az += anchor[n * 3 + 2]; }
  sx[tid] = ax; sy[tid] = ay; sz[tid] = az;
  __syncthreads();
  for (int st = 128; st > 0; st >>= 1) {
    if (tid < st) { sx[tid] += sx[tid + st]; sy[tid] += sy[tid + st]; sz[tid] += sz[tid + st]; }
    __syncthreads();
  }
  if (tid == 0) {
    float d = fmaxf((float)vc, 1.0f);
    stats[0] = sx[0] / d; stats[1] = sy[0] / d; stats[2] = sz[0] / d;
  }
}

__global__ void dist_kernel(const float* __restrict__ anchor,
                            const float* __restrict__ stats,
                            const int* __restrict__ vcp,
                            float* __restrict__ buf) {
  int n = blockIdx.x * 256 + threadIdx.x;
  if (n >= 4096) return;
  float dx = anchor[n * 3] - stats[0];
  float dy = anchor[n * 3 + 1] - stats[1];
  float dz = anchor[n * 3 + 2] - stats[2];
  buf[n] = (n < vcp[0]) ? sqrtf(dx * dx + dy * dy + dz * dz) : 3.0e38f;
}

__global__ __launch_bounds__(1024) void sortscale_kernel(
    const float* __restrict__ buf, const int* __restrict__ vcp,
    float* __restrict__ stats) {
  __shared__ float ss[4096];
  int tid = threadIdx.x;
  for (int i = tid; i < 4096; i += 1024) ss[i] = buf[i];
  __syncthreads();
  for (int k = 2; k <= 4096; k <<= 1) {
    for (int j = k >> 1; j > 0; j >>= 1) {
      for (int i = tid; i < 4096; i += 1024) {
        int ixj = i ^ j;
        if (ixj > i) {
          float a = ss[i], b = ss[ixj];
          bool up = ((i & k) == 0);
          bool sw = up ? (a > b) : (a < b);
          if (sw) { ss[i] = b; ss[ixj] = a; }
        }
      }
      __syncthreads();
    }
  }
  if (tid == 0) {
    int vc = vcp[0];
    float pos = 0.95f * ((float)vc - 1.0f);
    int lo = (int)floorf(pos);
    int hi = (int)ceilf(pos);
    if (lo < 0) lo = 0;
    if (hi > 4095) hi = 4095;
    float dlo = ss[lo], dhi = ss[hi];
    stats[3] = fmaxf(dlo + (pos - (float)lo) * (dhi - dlo), 1e-8f);
  }
}

__global__ void encode_kernel(const float* __restrict__ anchor,
                              const float* __restrict__ stats,
                              float* __restrict__ enc) {
  int n = blockIdx.x * 256 + threadIdx.x;
  if (n >= 4096) return;
  float inv = 1.0f / stats[3];
  float p[3];
  for (int j = 0; j < 3; ++j) p[j] = (anchor[n * 3 + j] - stats[j]) * inv;
  float* e = enc + (size_t)n * 39;
  e[0] = p[0]; e[1] = p[1]; e[2] = p[2];
  for (int ax = 0; ax < 3; ++ax)
    for (int f = 0; f < 6; ++f) {
      float xin = p[ax] * (float)(1 << f) * 3.14159265358979323846f;
      e[3 + ax * 6 + f]  = sinf(xin);
      e[21 + ax * 6 + f] = cosf(xin);
    }
}

// ---------------------------------------------------------------------------
// Projection + bilinear sampling. Grid (4096, 8), block 128 (channels).
// ---------------------------------------------------------------------------
__global__ __launch_bounds__(128) void sample_kernel(
    const float* __restrict__ anchor, const float* __restrict__ extr,
    const float* __restrict__ intr, const float* __restrict__ feat,
    float* __restrict__ samp, float* __restrict__ vis) {
  int n = blockIdx.x, v = blockIdx.y, c = threadIdx.x;
  float p0 = anchor[n * 3], p1 = anchor[n * 3 + 1], p2 = anchor[n * 3 + 2];
  const float* E = extr + v * 16;
  float c0 = E[0] * p0 + E[1] * p1 + E[2] * p2 + E[3];
  float c1 = E[4] * p0 + E[5] * p1 + E[6] * p2 + E[7];
  float c2 = E[8] * p0 + E[9] * p1 + E[10] * p2 + E[11];
  const float* Ki = intr + v * 9;
  float px = Ki[0] * c0 + Ki[1] * c1 + Ki[2] * c2;
  float py = Ki[3] * c0 + Ki[4] * c1 + Ki[5] * c2;
  float pz = Ki[6] * c0 + Ki[7] * c1 + Ki[8] * c2;
  float zs = (fabsf(pz) > 1e-6f) ? pz : 1e-6f;
  float u = px / zs, vv = py / zs;
  bool ok = (pz > 0.001f) && (u >= 0.0f) && (u <= 255.0f) && (vv >= 0.0f) && (vv <= 255.0f);
  float x = fminf(fmaxf(u, 0.0f), 255.0f);
  float y = fminf(fmaxf(vv, 0.0f), 255.0f);
  int x0 = (int)floorf(x), y0 = (int)floorf(y);
  int x1 = (x0 + 1 > 255) ? 255 : x0 + 1;
  int y1 = (y0 + 1 > 255) ? 255 : y0 + 1;
  float wx = x - (float)x0, wy = y - (float)y0;
  const float* F = feat + ((size_t)v * 128 + c) * 65536;
  float f00 = F[y0 * 256 + x0], f01 = F[y0 * 256 + x1];
  float f10 = F[y1 * 256 + x0], f11 = F[y1 * 256 + x1];
  samp[((size_t)n * 8 + v) * 128 + c] =
      f00 * (1 - wx) * (1 - wy) + f01 * wx * (1 - wy) +
      f10 * (1 - wx) * wy + f11 * wx * wy;
  if (c == 0) vis[n * 8 + v] = ok ? 1.0f : 0.0f;
}

// ---------------------------------------------------------------------------
// View softmax + aggregation + q_all assembly. One block per anchor.
// ---------------------------------------------------------------------------
__global__ __launch_bounds__(256) void viewagg_kernel(
    const float* __restrict__ hmid, const float* __restrict__ vw2,
    const float* __restrict__ vb2, const float* __restrict__ vis,
    const float* __restrict__ proj, const float* __restrict__ query,
    float* __restrict__ xa, const int* __restrict__ vcp) {
  int n = blockIdx.x;
  int tid = threadIdx.x, wave = tid >> 5, lane = tid & 31;
  __shared__ float wv[8];
  float p = 0.0f;
  for (int c = lane; c < 128; c += 32)
    p += hmid[((size_t)n * 8 + wave) * 128 + c] * vw2[c];
  for (int off = 16; off > 0; off >>= 1) p += __shfl_xor(p, off, 32);
  if (lane == 0)
    wv[wave] = p + vb2[0] + (vis[n * 8 + wave] > 0.5f ? 0.0f : -1.0e9f);
  __syncthreads();
  if (tid == 0) {
    float mx = wv[0];
    for (int v2 = 1; v2 < 8; ++v2) mx = fmaxf(mx, wv[v2]);
    float s = 0.0f, e[8];
    for (int v2 = 0; v2 < 8; ++v2) { e[v2] = expf(wv[v2] - mx); s += e[v2]; }
    for (int v2 = 0; v2 < 8; ++v2) wv[v2] = e[v2] / s;
  }
  __syncthreads();
  float mask = (n < vcp[0]) ? 1.0f : 0.0f;
  for (int c = tid; c < 256; c += 256) {
    float a = 0.0f;
    for (int v2 = 0; v2 < 8; ++v2)
      a += wv[v2] * proj[((size_t)n * 8 + v2) * 256 + c];
    xa[(size_t)n * 512 + c] = query[(size_t)n * 256 + c] * mask;
    xa[(size_t)n * 512 + 256 + c] = a * mask;
  }
}

// ---------------------------------------------------------------------------
// Slot feature assembly / offsets / opacity mask
// ---------------------------------------------------------------------------
__global__ __launch_bounds__(256) void slotfeat_kernel(
    const float* __restrict__ qlat, const float* __restrict__ slots,
    float* __restrict__ sf) {
  int idx = blockIdx.x * 256 + threadIdx.x;
  if (idx >= 16384 * 128) return;
  int row = idx >> 7, c = idx & 127;
  int n = row >> 2, s = row & 3;
  sf[idx] = (c < 64) ? qlat[(size_t)n * 64 + c] : slots[s * 64 + (c - 64)];
}

__global__ __launch_bounds__(256) void offsets_kernel(
    const float* __restrict__ offm, const float* __restrict__ w2,
    const float* __restrict__ b2, const float* __restrict__ vox,
    const float* __restrict__ anchor, float* __restrict__ gpos) {
  int row = blockIdx.x * 256 + threadIdx.x;
  if (row >= 16384) return;
  float a0 = b2[0], a1 = b2[1], a2 = b2[2];
  const float* h = offm + (size_t)row * 64;
  for (int k = 0; k < 64; ++k) {
    float hv = h[k];
    a0 += hv * w2[k * 3 + 0];
    a1 += hv * w2[k * 3 + 1];
    a2 += hv * w2[k * 3 + 2];
  }
  float vs = vox[0];
  int n = row >> 2;
  gpos[(size_t)row * 3 + 0] = anchor[n * 3 + 0] + tanhf(a0) * vs;
  gpos[(size_t)row * 3 + 1] = anchor[n * 3 + 1] + tanhf(a1) * vs;
  gpos[(size_t)row * 3 + 2] = anchor[n * 3 + 2] + tanhf(a2) * vs;
}

__global__ void opacity_kernel(float* __restrict__ gs, const int* __restrict__ vcp) {
  int row = blockIdx.x * 256 + threadIdx.x;
  if (row >= 16384) return;
  if ((row >> 2) >= vcp[0]) gs[(size_t)row * 85] = -20.0f;
}

// ---------------------------------------------------------------------------
// Workspace layout (floats)
// ---------------------------------------------------------------------------
static constexpr size_t O_STATS = 0;
static constexpr size_t O_VIS   = 16;
static constexpr size_t O_SORT  = O_VIS + 32768;
static constexpr size_t O_ENC   = O_SORT + 4096;
static constexpr size_t O_QUERY = O_ENC + 4096ULL * 39;
static constexpr size_t O_XA    = O_QUERY + 4096ULL * 256;
static constexpr size_t O_POOL  = O_XA + 4096ULL * 512;
// phase 1 (sampling/aggregation)
static constexpr size_t O_SAMP = O_POOL;
static constexpr size_t O_PROJ = O_SAMP + 32768ULL * 128;
static constexpr size_t O_HMID = O_PROJ + 32768ULL * 256;
static constexpr size_t O_QMID = O_HMID + 32768ULL * 128;
// phase 2 (transformer) -- aliases phase 1
static constexpr size_t O_H    = O_POOL;
static constexpr size_t O_QKV  = O_H + 4096ULL * 512;
static constexpr size_t O_O    = O_QKV + 4096ULL * 1536;
static constexpr size_t O_MID  = O_O + 4096ULL * 512;
// phase 3 (heads) -- aliases phase 1/2
static constexpr size_t O_LAT1  = O_POOL;
static constexpr size_t O_LAT1G = O_LAT1 + 4096ULL * 256;
static constexpr size_t O_LAT   = O_LAT1G + 4096ULL * 256;
static constexpr size_t O_SLOTF = O_LAT + 4096ULL * 64;
static constexpr size_t O_OFFM  = O_SLOTF + 16384ULL * 128;
static constexpr size_t O_OUTM  = O_OFFM + 16384ULL * 64;
// output offsets (floats)
static constexpr size_t GS_POS_OFF = 16384ULL * 85;
static constexpr size_t QLAT_OFF   = GS_POS_OFF + 16384ULL * 3;

extern "C" void kernel_launch(void* const* d_in, const int* in_sizes, int n_in,
                              void* d_out, int out_size, void* d_ws, size_t ws_size,
                              hipStream_t stream) {
  (void)in_sizes; (void)n_in; (void)out_size; (void)ws_size;
  const float* anchor = (const float*)d_in[0];
  const int*   vcp    = (const int*)d_in[1];
  const float* vox    = (const float*)d_in[2];
  const float* feat   = (const float*)d_in[3];
  const float* extr   = (const float*)d_in[4];
  const float* intr   = (const float*)d_in[5];
  const float* pe_w1 = (const float*)d_in[6];  const float* pe_b1 = (const float*)d_in[7];
  const float* pe_w2 = (const float*)d_in[8];  const float* pe_b2 = (const float*)d_in[9];
  const float* fp_w  = (const float*)d_in[10]; const float* fp_b  = (const float*)d_in[11];
  const float* va_w1 = (const float*)d_in[12]; const float* va_b1 = (const float*)d_in[13];
  const float* va_w2 = (const float*)d_in[14]; const float* va_b2 = (const float*)d_in[15];
  const float* ln1_w = (const float*)d_in[16]; const float* ln1_b = (const float*)d_in[17];
  const float* ln2_w = (const float*)d_in[18]; const float* ln2_b = (const float*)d_in[19];
  const float* qkv_w = (const float*)d_in[20]; const float* qkv_b = (const float*)d_in[21];
  const float* ao_w  = (const float*)d_in[22]; const float* ao_b  = (const float*)d_in[23];
  const float* m1_w  = (const float*)d_in[24]; const float* m1_b  = (const float*)d_in[25];
  const float* m2_w  = (const float*)d_in[26]; const float* m2_b  = (const float*)d_in[27];
  const float* lp_w1 = (const float*)d_in[28]; const float* lp_b1 = (const float*)d_in[29];
  const float* lp_lnw= (const float*)d_in[30]; const float* lp_lnb= (const float*)d_in[31];
  const float* lp_w2 = (const float*)d_in[32]; const float* lp_b2 = (const float*)d_in[33];
  const float* fln_w = (const float*)d_in[34]; const float* fln_b = (const float*)d_in[35];
  const float* out_w1= (const float*)d_in[36]; const float* out_b1= (const float*)d_in[37];
  const float* out_w2= (const float*)d_in[38]; const float* out_b2= (const float*)d_in[39];
  const float* off_w1= (const float*)d_in[40]; const float* off_b1= (const float*)d_in[41];
  const float* off_w2= (const float*)d_in[42]; const float* off_b2= (const float*)d_in[43];
  const float* slots = (const float*)d_in[44];

  float* ws  = (float*)d_ws;
  float* out = (float*)d_out;

  // ---- Phase 0: anchor stats + positional encoding ----
  center_kernel<<<1, 256, 0, stream>>>(anchor, vcp, ws + O_STATS);
  dist_kernel<<<16, 256, 0, stream>>>(anchor, ws + O_STATS, vcp, ws + O_SORT);
  sortscale_kernel<<<1, 1024, 0, stream>>>(ws + O_SORT, vcp, ws + O_STATS);
  encode_kernel<<<16, 256, 0, stream>>>(anchor, ws + O_STATS, ws + O_ENC);
  gemm_kernel<1><<<dim3(64, 2), 256, 0, stream>>>(ws + O_ENC, nullptr, 1,
      pe_w1, pe_b1, nullptr, ws + O_QMID, 4096, 256, 39);
  gemm_kernel<0><<<dim3(64, 2), 256, 0, stream>>>(ws + O_QMID, nullptr, 1,
      pe_w2, pe_b2, nullptr, ws + O_QUERY, 4096, 256, 256);

  // ---- Phase 1: sampling + view aggregation ----
  sample_kernel<<<dim3(4096, 8), 128, 0, stream>>>(anchor, extr, intr, feat,
      ws + O_SAMP, ws + O_VIS);
  gemm_kernel<0><<<dim3(512, 2), 256, 0, stream>>>(ws + O_SAMP, nullptr, 1,
      fp_w, fp_b, nullptr, ws + O_PROJ, 32768, 256, 128);
  gemm_kernel<1><<<dim3(512, 1), 256, 0, stream>>>(ws + O_PROJ, ws + O_QUERY, 8,
      va_w1, va_b1, nullptr, ws + O_HMID, 32768, 128, 256);
  viewagg_kernel<<<4096, 256, 0, stream>>>(ws + O_HMID, va_w2, va_b2,
      ws + O_VIS, ws + O_PROJ, ws + O_QUERY, ws + O_XA, vcp);

  // ---- Phase 2: transformer ----
  for (int l = 0; l < 4; ++l) {
    layernorm_kernel<<<4096, 256, 0, stream>>>(ws + O_XA, ln1_w + l * 512,
        ln1_b + l * 512, ws + O_H, 512, 0);
    gemm_kernel<0><<<dim3(64, 12), 256, 0, stream>>>(ws + O_H, nullptr, 1,
        qkv_w + (size_t)l * 512 * 1536, qkv_b + l * 1536, nullptr,
        ws + O_QKV, 4096, 1536, 512);
    attn_kernel<<<dim3(64, 8), 128, 0, stream>>>(ws + O_QKV, ws + O_O);
    gemm_kernel<0><<<dim3(64, 4), 256, 0, stream>>>(ws + O_O, nullptr, 1,
        ao_w + (size_t)l * 512 * 512, ao_b + l * 512, ws + O_XA,
        ws + O_XA, 4096, 512, 512);
    layernorm_kernel<<<4096, 256, 0, stream>>>(ws + O_XA, ln2_w + l * 512,
        ln2_b + l * 512, ws + O_H, 512, 0);
    gemm_kernel<1><<<dim3(64, 16), 256, 0, stream>>>(ws + O_H, nullptr, 1,
        m1_w + (size_t)l * 512 * 2048, m1_b + l * 2048, nullptr,
        ws + O_MID, 4096, 2048, 512);
    gemm_kernel<0><<<dim3(64, 4), 256, 0, stream>>>(ws + O_MID, nullptr, 1,
        m2_w + (size_t)l * 2048 * 512, m2_b + l * 512, ws + O_XA,
        ws + O_XA, 4096, 512, 2048);
  }

  // ---- Phase 3: latent + Gaussian heads ----
  gemm_kernel<0><<<dim3(64, 2), 256, 0, stream>>>(ws + O_XA, nullptr, 1,
      lp_w1, lp_b1, nullptr, ws + O_LAT1, 4096, 256, 512);
  layernorm_kernel<<<4096, 256, 0, stream>>>(ws + O_LAT1, lp_lnw, lp_lnb,
      ws + O_LAT1G, 256, 1);
  gemm_kernel<0><<<dim3(64, 1), 256, 0, stream>>>(ws + O_LAT1G, nullptr, 1,
      lp_w2, lp_b2, nullptr, ws + O_LAT, 4096, 64, 256);
  layernorm_kernel<<<4096, 256, 0, stream>>>(ws + O_LAT, fln_w, fln_b,
      out + QLAT_OFF, 64, 0);
  slotfeat_kernel<<<8192, 256, 0, stream>>>(out + QLAT_OFF, slots, ws + O_SLOTF);
  gemm_kernel<1><<<dim3(256, 1), 256, 0, stream>>>(ws + O_SLOTF, nullptr, 1,
      off_w1, off_b1, nullptr, ws + O_OFFM, 16384, 64, 128);
  offsets_kernel<<<64, 256, 0, stream>>>(ws + O_OFFM, off_w2, off_b2, vox,
      anchor, out + GS_POS_OFF);
  gemm_kernel<1><<<dim3(256, 1), 256, 0, stream>>>(ws + O_SLOTF, nullptr, 1,
      out_w1, out_b1, nullptr, ws + O_OUTM, 16384, 128, 128);
  gemm_kernel<0><<<dim3(256, 1), 256, 0, stream>>>(ws + O_OUTM, nullptr, 1,
      out_w2, out_b2, nullptr, out, 16384, 85, 128);
  opacity_kernel<<<64, 256, 0, stream>>>(out, vcp);
}
